// MLPKANDecoder_84628035601291
// MI455X (gfx1250) — compile-verified, matrix operand order checked by reading, not run on hardware
//
#include <hip/hip_runtime.h>
#include <cstdint>

typedef __bf16 bf16;
typedef __attribute__((ext_vector_type(16))) __bf16 v16bf;
typedef __attribute__((ext_vector_type(8)))  __bf16 v8bf;
typedef __attribute__((ext_vector_type(8)))  float   v8f;

union BF16x16 { v16bf v; v8bf h[2]; };
union BF16x8pk { bf16 h[8]; uint4 u; };

#define LDH 40   // LDS row stride in bf16 elements (80B = 5*16B): conflict-free b128 reads

// raw 32-bit LDS offset of a __shared__ pointer (addrspace(3) cast)
#define LDS_OFF(p) ((unsigned)(uintptr_t)(__attribute__((address_space(3))) const void*)(p))

__device__ __forceinline__ float silu_f(float x) { return x / (1.0f + __expf(-x)); }

// ---------------------------------------------------------------------------
// Head weight fusion: Wf[o][c] = (wr[:,64i:64i+64] @ w_i)[o][c-base_i]  (bf16)
// ---------------------------------------------------------------------------
__global__ void k_fuse_w(const float* __restrict__ wr,
                         const float* __restrict__ w1, const float* __restrict__ w2,
                         const float* __restrict__ w3, const float* __restrict__ w4,
                         bf16* __restrict__ Wf) {
  int idx = blockIdx.x * 256 + threadIdx.x;           // 64*512
  int o = idx >> 9, c = idx & 511;
  const float* wsrc; int cin, coff, br;
  if (c < 32)       { wsrc = w1; cin = 32;  coff = c;       br = 0; }
  else if (c < 96)  { wsrc = w2; cin = 64;  coff = c - 32;  br = 1; }
  else if (c < 256) { wsrc = w3; cin = 160; coff = c - 96;  br = 2; }
  else              { wsrc = w4; cin = 256; coff = c - 256; br = 3; }
  float acc = 0.0f;
  for (int j = 0; j < 64; ++j)
    acc += wr[o * 256 + br * 64 + j] * wsrc[j * cin + coff];
  Wf[o * 512 + c] = (bf16)acc;
}

__global__ void k_fuse_b(const float* __restrict__ wr,
                         const float* __restrict__ b1, const float* __restrict__ b2,
                         const float* __restrict__ b3, const float* __restrict__ b4,
                         const float* __restrict__ brr, float* __restrict__ bfu) {
  int o = threadIdx.x;                                 // 64
  float acc = brr[o];
  const float* bs[4] = { b1, b2, b3, b4 };
  for (int i = 0; i < 4; ++i)
    for (int j = 0; j < 64; ++j)
      acc += wr[o * 256 + i * 64 + j] * bs[i][j];
  bfu[o] = acc;
}

// ---------------------------------------------------------------------------
// Bilinear (align_corners) resample of raw inputs into concat bf16 (b,p,c)
// (resize commutes with the 1x1 convs, which were folded into Wf)
// ---------------------------------------------------------------------------
__global__ void k_interp(const float* __restrict__ x1, const float* __restrict__ x2,
                         const float* __restrict__ x3, const float* __restrict__ x4,
                         bf16* __restrict__ XCt) {
  int idx = blockIdx.x * 256 + threadIdx.x;            // 256*64*512
  int c = idx & 511;
  int p = (idx >> 9) & 63;
  int b = idx >> 15;
  int h8 = p >> 3, w8 = p & 7;
  const float* src; int cin, cl, S;
  if (c < 32)       { src = x1; cin = 32;  cl = c;       S = 16; }
  else if (c < 96)  { src = x2; cin = 64;  cl = c - 32;  S = 8;  }
  else if (c < 256) { src = x3; cin = 160; cl = c - 96;  S = 4;  }
  else              { src = x4; cin = 256; cl = c - 256; S = 2;  }
  float scale = (float)(S - 1) / 7.0f;
  float fy = h8 * scale; int iy = (int)fy; if (iy > S - 2) iy = S - 2; fy -= (float)iy;
  float fx = w8 * scale; int ix = (int)fx; if (ix > S - 2) ix = S - 2; fx -= (float)ix;
  const float* base = src + (size_t)(b * cin + cl) * S * S;
  float v00 = base[iy * S + ix],       v01 = base[iy * S + ix + 1];
  float v10 = base[(iy + 1) * S + ix], v11 = base[(iy + 1) * S + ix + 1];
  float v = (1.0f - fy) * ((1.0f - fx) * v00 + fx * v01)
          +         fy  * ((1.0f - fx) * v10 + fx * v11);
  XCt[(size_t)(b * 64 + p) * 512 + c] = (bf16)v;
}

// ---------------------------------------------------------------------------
// Head GEMM: xf[b, o*64+p] = XCt[(b,p),:] @ Wf[o,:] + bfused[o]
// M=16384 (b,p), N=64, K=512.  BM=256, BN=64, BK=32, 8 waves (4M x 2N).
// ---------------------------------------------------------------------------
__global__ void __launch_bounds__(256) k_head_gemm(const bf16* __restrict__ A,
                                                   const bf16* __restrict__ Wf,
                                                   const float* __restrict__ bfu,
                                                   float* __restrict__ xf) {
  __shared__ bf16 lA[256 * LDH];
  __shared__ bf16 lB[64 * LDH];
  int tid = threadIdx.x, lane = tid & 31, wave = tid >> 5;
  int wm = wave >> 1, wn = wave & 1;
  int m0 = blockIdx.x * 256;
  v8f acc[4][2] = {};
  int mrow = wm * 64 + (lane & 15);
  int nrow = wn * 32 + (lane & 15);
  int hsel = (lane >> 4) * 8;
  int oB = tid >> 2, qB = tid & 3;

  for (int k0 = 0; k0 < 512; k0 += 32) {
    { // A tile: thread -> one row, 64B
      const uint4* g = (const uint4*)(A + (size_t)(m0 + tid) * 512 + k0);
      uint4 d0 = g[0], d1 = g[1], d2 = g[2], d3 = g[3];
      *(uint4*)&lA[tid * LDH +  0] = d0;
      *(uint4*)&lA[tid * LDH +  8] = d1;
      *(uint4*)&lA[tid * LDH + 16] = d2;
      *(uint4*)&lA[tid * LDH + 24] = d3;
    }
    { // B tile: bf16 weights, 16B per thread
      uint4 d = *(const uint4*)(Wf + (size_t)oB * 512 + k0 + qB * 8);
      *(uint4*)&lB[oB * LDH + qB * 8] = d;
    }
    __syncthreads();
    BF16x16 bfr[2];
#pragma unroll
    for (int sn = 0; sn < 2; ++sn) {
      int nb = (nrow + sn * 16) * LDH + hsel;
      bfr[sn].h[0] = *(const v8bf*)&lB[nb];
      bfr[sn].h[1] = *(const v8bf*)&lB[nb + 16];
    }
#pragma unroll
    for (int sm = 0; sm < 4; ++sm) {
      int mb = (mrow + sm * 16) * LDH + hsel;
      BF16x16 af;
      af.h[0] = *(const v8bf*)&lA[mb];
      af.h[1] = *(const v8bf*)&lA[mb + 16];
#pragma unroll
      for (int sn = 0; sn < 2; ++sn)
        acc[sm][sn] = __builtin_amdgcn_wmma_f32_16x16x32_bf16(
            false, af.v, false, bfr[sn].v, (short)0, acc[sm][sn], false, false);
    }
    __syncthreads();
  }
#pragma unroll
  for (int sm = 0; sm < 4; ++sm)
#pragma unroll
    for (int sn = 0; sn < 2; ++sn) {
      int col = wn * 32 + sn * 16 + (lane & 15);
      float bias = bfu[col];
#pragma unroll
      for (int r = 0; r < 8; ++r) {
        int grow = m0 + wm * 64 + sm * 16 + r + ((lane >> 4) << 3);
        int b = grow >> 6, p = grow & 63;
        xf[(size_t)b * 4096 + col * 64 + p] = acc[sm][sn][r] + bias;
      }
    }
}

// ---------------------------------------------------------------------------
// KAN prepare: Abuf[b] = [ silu(h) (in) | bsplines(h) (6*in) ]  in bf16
// Uniform cubic B-splines on grid [-1,1], GS=3, K=3 (all denoms = k*h).
// ---------------------------------------------------------------------------
__global__ void k_prepare(const float* __restrict__ H, bf16* __restrict__ Ab,
                          int in, int inShift) {
  int idx = blockIdx.x * 256 + threadIdx.x;            // 256*in
  int b = idx >> inShift;
  int i = idx & (in - 1);
  float x = H[idx];
  size_t rb = (size_t)b * 7 * in;
  Ab[rb + i] = (bf16)silu_f(x);
  const float hh = 2.0f / 3.0f;
  float g[10];
#pragma unroll
  for (int j = 0; j < 10; ++j) g[j] = (float)(j - 3) * hh - 1.0f;
  float bb[9];
#pragma unroll
  for (int j = 0; j < 9; ++j) bb[j] = (x >= g[j] && x < g[j + 1]) ? 1.0f : 0.0f;
#pragma unroll
  for (int k = 1; k <= 3; ++k) {
    float inv = 1.0f / ((float)k * hh);
#pragma unroll
    for (int j = 0; j <= 8 - k; ++j)
      bb[j] = ((x - g[j]) * bb[j] + (g[j + k + 1] - x) * bb[j + 1]) * inv;
  }
  bf16* sp = Ab + rb + in + (size_t)i * 6;
#pragma unroll
  for (int j = 0; j < 6; ++j) sp[j] = (bf16)bb[j];
}

// ---------------------------------------------------------------------------
// KAN GEMM: C = A @ [bw | sw*sc]^T.  M=256 (full batch), BN=64, BK=32,
// split-K=8 into deterministic partial buffers.
// A tile: async global->LDS DMA (ASYNCcnt path).  B tile: fp32 weights
// streamed through VGPRs, sc scaler fused, converted to bf16.
// ---------------------------------------------------------------------------
__global__ void __launch_bounds__(256) k_kan_gemm(const bf16* __restrict__ A,
                                                  const float* __restrict__ bw,
                                                  const float* __restrict__ sw,
                                                  const float* __restrict__ sc,
                                                  float* __restrict__ Cpart,
                                                  int N, int Kin, int itersPerSplit) {
  __shared__ bf16 lA[256 * LDH];
  __shared__ bf16 lB[64 * LDH];
  int tid = threadIdx.x, lane = tid & 31, wave = tid >> 5;
  int wm = wave >> 1, wn = wave & 1;
  int n0 = blockIdx.x * 64;
  int Ktot = 7 * Kin;
  int it0 = blockIdx.y * itersPerSplit;
  v8f acc[4][2] = {};
  int mrow = wm * 64 + (lane & 15);
  int nrow = wn * 32 + (lane & 15);
  int hsel = (lane >> 4) * 8;
  int oRow = n0 + (tid >> 2);
  int qB = tid & 3;
  unsigned aLds = LDS_OFF(&lA[tid * LDH]);   // per-thread LDS row offset (16B aligned: 80*t)

  for (int it = it0; it < it0 + itersPerSplit; ++it) {
    int k0 = it * 32;
    { // A tile: thread -> row tid, 64B, via async DMA to LDS.
      // Same inst-offset is applied to both LDS and global addresses (ISA 8.4.4).
      const bf16* gA = A + (size_t)tid * Ktot + k0;
      asm volatile(
          "global_load_async_to_lds_b128 %0, %1, off\n\t"
          "global_load_async_to_lds_b128 %0, %1, off offset:16\n\t"
          "global_load_async_to_lds_b128 %0, %1, off offset:32\n\t"
          "global_load_async_to_lds_b128 %0, %1, off offset:48"
          :: "v"(aLds), "v"(gA) : "memory");
    }
    { // B tile: 8 logical k's per thread; fp32 -> bf16 with sc fusion
      int kk = k0 + qB * 8;
      float vals[8];
      if (kk < Kin) {
        const float4* g = (const float4*)(bw + (size_t)oRow * Kin + kk);
        float4 a = g[0], b2 = g[1];
        vals[0] = a.x;  vals[1] = a.y;  vals[2] = a.z;  vals[3] = a.w;
        vals[4] = b2.x; vals[5] = b2.y; vals[6] = b2.z; vals[7] = b2.w;
      } else {
        int ks = kk - Kin;                      // spline-region flat index
        const float4* g = (const float4*)(sw + (size_t)oRow * 6 * Kin + ks);
        float4 a = g[0], b2 = g[1];
        float sv[8] = { a.x, a.y, a.z, a.w, b2.x, b2.y, b2.z, b2.w };
        const float* scp = sc + (size_t)oRow * Kin;
#pragma unroll
        for (int j = 0; j < 8; ++j) vals[j] = sv[j] * scp[(ks + j) / 6];
      }
      // prefetch next K-tile of the weight stream (global_prefetch_b8)
      int kn = k0 + 32 + qB * 8;
      if (kn < Ktot) {
        const void* pf = (kn < Kin) ? (const void*)(bw + (size_t)oRow * Kin + kn)
                                    : (const void*)(sw + (size_t)oRow * 6 * Kin + (kn - Kin));
        __builtin_prefetch(pf, 0, 1);
      }
      BF16x8pk pk;
#pragma unroll
      for (int j = 0; j < 8; ++j) pk.h[j] = (bf16)vals[j];
      *(uint4*)&lB[(tid >> 2) * LDH + qB * 8] = pk.u;
    }
    // all async A-tile writes must land in LDS before any wave reads the tile
    asm volatile("s_wait_asynccnt 0x0" ::: "memory");
    __syncthreads();
    BF16x16 bfr[2];
#pragma unroll
    for (int sn = 0; sn < 2; ++sn) {
      int nb = (nrow + sn * 16) * LDH + hsel;
      bfr[sn].h[0] = *(const v8bf*)&lB[nb];
      bfr[sn].h[1] = *(const v8bf*)&lB[nb + 16];
    }
#pragma unroll
    for (int sm = 0; sm < 4; ++sm) {
      int mb = (mrow + sm * 16) * LDH + hsel;
      BF16x16 af;
      af.h[0] = *(const v8bf*)&lA[mb];
      af.h[1] = *(const v8bf*)&lA[mb + 16];
#pragma unroll
      for (int sn = 0; sn < 2; ++sn)
        acc[sm][sn] = __builtin_amdgcn_wmma_f32_16x16x32_bf16(
            false, af.v, false, bfr[sn].v, (short)0, acc[sm][sn], false, false);
    }
    __syncthreads();
  }
  // deterministic split-K: plain stores into per-split partial buffer
  float* Cp = Cpart + (size_t)blockIdx.y * 256 * N;
#pragma unroll
  for (int sm = 0; sm < 4; ++sm)
#pragma unroll
    for (int sn = 0; sn < 2; ++sn) {
      int col = n0 + wn * 32 + sn * 16 + (lane & 15);
#pragma unroll
      for (int r = 0; r < 8; ++r) {
        int row = wm * 64 + sm * 16 + r + ((lane >> 4) << 3);
        Cp[(size_t)row * N + col] = acc[sm][sn][r];
      }
    }
}

__global__ void k_reduce8(const float* __restrict__ P, float* __restrict__ H, int n) {
  int idx = blockIdx.x * 256 + threadIdx.x;
  if (idx >= n) return;
  float a = 0.0f;
#pragma unroll
  for (int s = 0; s < 8; ++s) a += P[(size_t)s * n + idx];
  H[idx] = a;
}

// ---------------------------------------------------------------------------
// Tail: BN(eval)+SiLU, 19x64 conv, bilinear 8->64 upsample
// ---------------------------------------------------------------------------
__global__ void k_bn_silu(const float* __restrict__ H, const float* __restrict__ ga,
                          const float* __restrict__ be, const float* __restrict__ mu,
                          const float* __restrict__ va, float* __restrict__ S) {
  int idx = blockIdx.x * 256 + threadIdx.x;            // 256*4096
  int c = (idx >> 6) & 63;
  float x = (H[idx] - mu[c]) * (rsqrtf(va[c] + 1e-5f) * ga[c]) + be[c];
  S[idx] = silu_f(x);
}

__global__ void k_zconv(const float* __restrict__ S, const float* __restrict__ wc,
                        const float* __restrict__ bc, float* __restrict__ Z) {
  int idx = blockIdx.x * 256 + threadIdx.x;            // 256*19*64
  if (idx >= 256 * 19 * 64) return;
  int p = idx & 63;
  int o = (idx >> 6) % 19;
  int b = idx / (19 * 64);
  float acc = bc[o];
  const float* sb = S + (size_t)b * 4096 + p;
  for (int c = 0; c < 64; ++c) acc += wc[o * 64 + c] * sb[c * 64];
  Z[idx] = acc;
}

__global__ void k_resize_out(const float* __restrict__ Z, float* __restrict__ out) {
  int idx = blockIdx.x * 256 + threadIdx.x;            // 256*19*64*64
  int pw = idx & 63;
  int ph = (idx >> 6) & 63;
  int o  = (idx >> 12) % 19;
  int b  = idx / (19 * 4096);
  const float sc7 = 7.0f / 63.0f;
  float fy = ph * sc7; int iy = (int)fy; if (iy > 6) iy = 6; fy -= (float)iy;
  float fx = pw * sc7; int ix = (int)fx; if (ix > 6) ix = 6; fx -= (float)ix;
  const float* zb = Z + ((size_t)b * 19 + o) * 64;
  float v00 = zb[iy * 8 + ix],     v01 = zb[iy * 8 + ix + 1];
  float v10 = zb[iy * 8 + 8 + ix], v11 = zb[iy * 8 + 8 + ix + 1];
  out[idx] = (1.0f - fy) * ((1.0f - fx) * v00 + fx * v01)
           +         fy  * ((1.0f - fx) * v10 + fx * v11);
}

// ---------------------------------------------------------------------------
extern "C" void kernel_launch(void* const* d_in, const int* in_sizes, int n_in,
                              void* d_out, int out_size, void* d_ws, size_t ws_size,
                              hipStream_t stream) {
  (void)in_sizes; (void)n_in; (void)out_size; (void)ws_size;
  const float* x1 = (const float*)d_in[0];
  const float* x2 = (const float*)d_in[1];
  const float* x3 = (const float*)d_in[2];
  const float* x4 = (const float*)d_in[3];
  const float* w1 = (const float*)d_in[4];  const float* b1 = (const float*)d_in[5];
  const float* w2 = (const float*)d_in[6];  const float* b2 = (const float*)d_in[7];
  const float* w3 = (const float*)d_in[8];  const float* b3 = (const float*)d_in[9];
  const float* w4 = (const float*)d_in[10]; const float* b4 = (const float*)d_in[11];
  const float* wr = (const float*)d_in[12]; const float* br = (const float*)d_in[13];
  const float* bw0 = (const float*)d_in[14]; const float* sw0 = (const float*)d_in[15]; const float* sc0 = (const float*)d_in[16];
  const float* bw1 = (const float*)d_in[17]; const float* sw1 = (const float*)d_in[18]; const float* sc1 = (const float*)d_in[19];
  const float* bw2 = (const float*)d_in[20]; const float* sw2 = (const float*)d_in[21]; const float* sc2 = (const float*)d_in[22];
  const float* bng = (const float*)d_in[23]; const float* bnb = (const float*)d_in[24];
  const float* bnm = (const float*)d_in[25]; const float* bnv = (const float*)d_in[26];
  const float* wc  = (const float*)d_in[27]; const float* bc  = (const float*)d_in[28];
  float* out = (float*)d_out;

  char* wp = (char*)d_ws;
  auto alloc = [&](size_t bytes) {
    void* p = (void*)wp;
    wp += (bytes + 255) & ~(size_t)255;
    return p;
  };
  bf16*  Wf  = (bf16*) alloc((size_t)64 * 512 * sizeof(bf16));
  float* bfu = (float*)alloc((size_t)64 * sizeof(float));
  bf16*  XCt = (bf16*) alloc((size_t)256 * 64 * 512 * sizeof(bf16));
  float* xf  = (float*)alloc((size_t)256 * 4096 * sizeof(float));
  bf16*  Ab  = (bf16*) alloc((size_t)256 * 7 * 4096 * sizeof(bf16));
  float* H1  = (float*)alloc((size_t)256 * 1024 * sizeof(float));
  float* H2  = (float*)alloc((size_t)256 * 1024 * sizeof(float));
  float* H3  = (float*)alloc((size_t)256 * 4096 * sizeof(float));
  float* Sb  = (float*)alloc((size_t)256 * 4096 * sizeof(float));
  float* Zb  = (float*)alloc((size_t)256 * 19 * 64 * sizeof(float));
  float* Pb  = (float*)alloc((size_t)8 * 256 * 4096 * sizeof(float));  // split-K partials (max layer)

  // Head
  k_fuse_w<<<128, 256, 0, stream>>>(wr, w1, w2, w3, w4, Wf);
  k_fuse_b<<<1, 64, 0, stream>>>(wr, b1, b2, b3, b4, br, bfu);
  k_interp<<<(256 * 64 * 512) / 256, 256, 0, stream>>>(x1, x2, x3, x4, XCt);
  k_head_gemm<<<64, 256, 0, stream>>>(XCt, Wf, bfu, xf);

  // KAN layer 0: 4096 -> 1024
  k_prepare<<<(256 * 4096) / 256, 256, 0, stream>>>(xf, Ab, 4096, 12);
  k_kan_gemm<<<dim3(1024 / 64, 8), 256, 0, stream>>>(Ab, bw0, sw0, sc0, Pb, 1024, 4096, 112);
  k_reduce8<<<(256 * 1024) / 256, 256, 0, stream>>>(Pb, H1, 256 * 1024);

  // KAN layer 1: 1024 -> 1024
  k_prepare<<<(256 * 1024) / 256, 256, 0, stream>>>(H1, Ab, 1024, 10);
  k_kan_gemm<<<dim3(1024 / 64, 8), 256, 0, stream>>>(Ab, bw1, sw1, sc1, Pb, 1024, 1024, 28);
  k_reduce8<<<(256 * 1024) / 256, 256, 0, stream>>>(Pb, H2, 256 * 1024);

  // KAN layer 2: 1024 -> 4096
  k_prepare<<<(256 * 1024) / 256, 256, 0, stream>>>(H2, Ab, 1024, 10);
  k_kan_gemm<<<dim3(4096 / 64, 8), 256, 0, stream>>>(Ab, bw2, sw2, sc2, Pb, 4096, 1024, 28);
  k_reduce8<<<(256 * 4096) / 256, 256, 0, stream>>>(Pb, H3, 256 * 4096);

  // Tail
  k_bn_silu<<<(256 * 4096) / 256, 256, 0, stream>>>(H3, bng, bnb, bnm, bnv, Sb);
  k_zconv<<<(256 * 19 * 64 + 255) / 256, 256, 0, stream>>>(Sb, wc, bc, Zb);
  k_resize_out<<<(256 * 19 * 64 * 64) / 256, 256, 0, stream>>>(Zb, out);
}
